// BestRQModel_19490561589310
// MI455X (gfx1250) — matrix operand order repaired and checked by reading, not compiled, not run on hardware
//
#include <hip/hip_runtime.h>

// Problem constants (B=16, T=512 -> BT=8192; D=256; E=16; N=8192; C=1)
#define BT 8192
#define DD 256
#define EE 16
#define NN 8192

typedef __attribute__((ext_vector_type(16))) __bf16 v16bf;
typedef __attribute__((ext_vector_type(8)))  float  v8f;

__device__ __forceinline__ unsigned short f2bf(float f) {
  unsigned int u = __float_as_uint(f);
  u += 0x7FFFu + ((u >> 16) & 1u);   // round-to-nearest-even
  return (unsigned short)(u >> 16);
}

union Frag16 { uint4 q[2]; unsigned short us[16]; v16bf v; };

// ---------------------------------------------------------------------------
// Kernel 1: LayerNorm + random projection.  One wave per token (wave32).
// proj[t, 0..15] = LN(xs[t,:]) @ P[:,0..15]
// ---------------------------------------------------------------------------
__global__ void k_lnproj(const float* __restrict__ xs, const float* __restrict__ gam,
                         const float* __restrict__ bet, const float* __restrict__ P,
                         float* __restrict__ proj, float* __restrict__ ce_sum) {
  if (blockIdx.x == 0 && threadIdx.x == 0) ce_sum[0] = 0.0f;  // stream-ordered zeroing
  int wid = threadIdx.x >> 5, lane = threadIdx.x & 31;
  int t = blockIdx.x * 8 + wid;
  if (t >= BT) return;
  const float* xr = xs + (size_t)t * DD;
  int d0 = lane * 8;
  float4 a0 = *(const float4*)(xr + d0);
  float4 a1 = *(const float4*)(xr + d0 + 4);
  float x[8] = {a0.x, a0.y, a0.z, a0.w, a1.x, a1.y, a1.z, a1.w};
  float s = 0.f, ss = 0.f;
#pragma unroll
  for (int j = 0; j < 8; ++j) { s += x[j]; ss += x[j] * x[j]; }
#pragma unroll
  for (int off = 16; off >= 1; off >>= 1) {
    s += __shfl_xor(s, off);
    ss += __shfl_xor(ss, off);
  }
  float mu = s * (1.0f / DD);
  float var = ss * (1.0f / DD) - mu * mu;
  float rstd = rsqrtf(var + 1e-5f);
  float4 g0 = *(const float4*)(gam + d0), g1 = *(const float4*)(gam + d0 + 4);
  float4 b0 = *(const float4*)(bet + d0), b1 = *(const float4*)(bet + d0 + 4);
  float gg[8] = {g0.x, g0.y, g0.z, g0.w, g1.x, g1.y, g1.z, g1.w};
  float bb[8] = {b0.x, b0.y, b0.z, b0.w, b1.x, b1.y, b1.z, b1.w};
  float h[8];
#pragma unroll
  for (int j = 0; j < 8; ++j) h[j] = (x[j] - mu) * rstd * gg[j] + bb[j];
  float acc[16];
#pragma unroll
  for (int e = 0; e < 16; ++e) acc[e] = 0.f;
#pragma unroll
  for (int j = 0; j < 8; ++j) {
    const float* pr = P + (size_t)(d0 + j) * EE;
    float4 p0 = *(const float4*)(pr + 0);
    float4 p1 = *(const float4*)(pr + 4);
    float4 p2 = *(const float4*)(pr + 8);
    float4 p3 = *(const float4*)(pr + 12);
    acc[0]  = fmaf(h[j], p0.x, acc[0]);  acc[1]  = fmaf(h[j], p0.y, acc[1]);
    acc[2]  = fmaf(h[j], p0.z, acc[2]);  acc[3]  = fmaf(h[j], p0.w, acc[3]);
    acc[4]  = fmaf(h[j], p1.x, acc[4]);  acc[5]  = fmaf(h[j], p1.y, acc[5]);
    acc[6]  = fmaf(h[j], p1.z, acc[6]);  acc[7]  = fmaf(h[j], p1.w, acc[7]);
    acc[8]  = fmaf(h[j], p2.x, acc[8]);  acc[9]  = fmaf(h[j], p2.y, acc[9]);
    acc[10] = fmaf(h[j], p2.z, acc[10]); acc[11] = fmaf(h[j], p2.w, acc[11]);
    acc[12] = fmaf(h[j], p3.x, acc[12]); acc[13] = fmaf(h[j], p3.y, acc[13]);
    acc[14] = fmaf(h[j], p3.z, acc[14]); acc[15] = fmaf(h[j], p3.w, acc[15]);
  }
#pragma unroll
  for (int off = 16; off >= 1; off >>= 1) {
#pragma unroll
    for (int e = 0; e < 16; ++e) acc[e] += __shfl_xor(acc[e], off);
  }
  if (lane == 0) {
    float* po = proj + (size_t)t * EE;
    *(float4*)(po + 0)  = make_float4(acc[0], acc[1], acc[2], acc[3]);
    *(float4*)(po + 4)  = make_float4(acc[4], acc[5], acc[6], acc[7]);
    *(float4*)(po + 8)  = make_float4(acc[8], acc[9], acc[10], acc[11]);
    *(float4*)(po + 12) = make_float4(acc[12], acc[13], acc[14], acc[15]);
  }
}

// ---------------------------------------------------------------------------
// Kernel 2: embedding squared norms  enorm[n] = sum_e emb[e,n]^2
// ---------------------------------------------------------------------------
__global__ void k_enorm(const float* __restrict__ emb, float* __restrict__ enorm) {
  int n = blockIdx.x * blockDim.x + threadIdx.x;
  if (n >= NN) return;
  float s = 0.f;
#pragma unroll
  for (int e = 0; e < EE; ++e) {
    float v = emb[(size_t)e * NN + n];
    s = fmaf(v, v, s);
  }
  enorm[n] = s;
}

// ---------------------------------------------------------------------------
// Kernel 3: nearest-codebook argmin per token (exact f32; one wave per token).
// score(n) = |e_n|^2 - 2 p.e_n  (|p|^2 constant per token, dropped)
// ---------------------------------------------------------------------------
__global__ void k_argmin(const float* __restrict__ proj, const float* __restrict__ emb,
                         const float* __restrict__ enorm, int* __restrict__ tgt) {
  int wid = threadIdx.x >> 5, lane = threadIdx.x & 31;
  int t = blockIdx.x * 8 + wid;
  if (t >= BT) return;
  const float* pr = proj + (size_t)t * EE;
  float4 p0 = *(const float4*)(pr + 0);
  float4 p1 = *(const float4*)(pr + 4);
  float4 p2 = *(const float4*)(pr + 8);
  float4 p3 = *(const float4*)(pr + 12);
  float best = 3.4e38f;
  int bestn = 0;
  for (int n = lane; n < NN; n += 32) {
    const float* eb = emb + n;
    float d = 0.f;
    d = fmaf(eb[0 * NN],  p0.x, d); d = fmaf(eb[1 * NN],  p0.y, d);
    d = fmaf(eb[2 * NN],  p0.z, d); d = fmaf(eb[3 * NN],  p0.w, d);
    d = fmaf(eb[4 * NN],  p1.x, d); d = fmaf(eb[5 * NN],  p1.y, d);
    d = fmaf(eb[6 * NN],  p1.z, d); d = fmaf(eb[7 * NN],  p1.w, d);
    d = fmaf(eb[8 * NN],  p2.x, d); d = fmaf(eb[9 * NN],  p2.y, d);
    d = fmaf(eb[10 * NN], p2.z, d); d = fmaf(eb[11 * NN], p2.w, d);
    d = fmaf(eb[12 * NN], p3.x, d); d = fmaf(eb[13 * NN], p3.y, d);
    d = fmaf(eb[14 * NN], p3.z, d); d = fmaf(eb[15 * NN], p3.w, d);
    float sc = enorm[n] - 2.0f * d;
    if (sc < best) { best = sc; bestn = n; }   // ascending n -> first min per lane
  }
#pragma unroll
  for (int off = 16; off >= 1; off >>= 1) {
    float ob = __shfl_xor(best, off);
    int on = __shfl_xor(bestn, off);
    if (ob < best || (ob == best && on < bestn)) { best = ob; bestn = on; }
  }
  if (lane == 0) tgt[t] = bestn;
}

// ---------------------------------------------------------------------------
// Kernel 4: pack top_n_out (f32 [D,N]) into bf16 B-fragments in this
// file's contiguous-k B layout (k=16h+s, same as A): dword j of lane l in tile(nt), chunk(kc)
// holds {W[kb+ko+2j, n], W[kb+ko+2j+1, n]}, kb=32*kc, ko=(l<16?0:16),
// n = 16*nt + (l&15).  One thread per output dword.
// ---------------------------------------------------------------------------
__global__ void k_wpack(const float* __restrict__ W, unsigned int* __restrict__ wsB) {
  int idx = blockIdx.x * blockDim.x + threadIdx.x;   // < D*N/2 = 1048576
  if (idx >= (DD * NN) / 2) return;
  int j = idx & 7;
  int lane = (idx >> 3) & 31;
  int kc = (idx >> 8) & 7;
  int nt = idx >> 11;
  int d0 = kc * 32 + ((lane & 16) ? 16 : 0) + 2 * j;
  int n = nt * 16 + (lane & 15);
  unsigned int lo = f2bf(W[(size_t)d0 * NN + n]);
  unsigned int hi = f2bf(W[(size_t)(d0 + 1) * NN + n]);
  wsB[idx] = lo | (hi << 16);
}

// ---------------------------------------------------------------------------
// Kernel 5: WMMA logits + streamed softmax + CE.  One block per 16-token
// M-tile; 8 waves sweep the 512 N-tiles, two tiles per iteration (dual
// accumulator chains to hide L2 latency behind the XDL pipe).
// ---------------------------------------------------------------------------
#define AROW 264   // padded LDS row stride (ushorts) -> conflict-free b128 reads
__global__ void __launch_bounds__(256)
k_main(const float* __restrict__ xs, const unsigned char* __restrict__ mmask,
       const float* __restrict__ mask_emb, const unsigned int* __restrict__ wsB,
       const int* __restrict__ tgt, const unsigned char* __restrict__ pad,
       float* __restrict__ ce_sum) {
  __shared__ unsigned short sA[16 * AROW];
  __shared__ int sTgt[16];
  __shared__ float sS[8][16];
  __shared__ float sTL[8][16];

  int tid = threadIdx.x;
  int t0 = blockIdx.x * 16;

  // Build masked_xs tile in bf16: 16 tokens x 256 dims
  {
    int tt = tid >> 4;
    int dbase = (tid & 15) * 16;
    bool mm = mmask[t0 + tt] != 0;
    const float* xr = xs + (size_t)(t0 + tt) * DD;
#pragma unroll
    for (int i = 0; i < 16; ++i) {
      int d = dbase + i;
      float v = mm ? mask_emb[d] : xr[d];
      sA[tt * AROW + d] = f2bf(v);
    }
  }
  if (tid < 16) sTgt[tid] = tgt[t0 + tid];
  __syncthreads();

  int wid = tid >> 5, lane = tid & 31;
  int trow = lane & 15;
  int koffA = (lane & 16) ? 16 : 0;
  int nc = lane & 15;
  int rbase = (lane & 16) ? 8 : 0;

  // Preload the 8 A fragments (whole K=256) for this lane's token row.
  Frag16 a[8];
#pragma unroll
  for (int kc = 0; kc < 8; ++kc) {
    const uint4* pA = (const uint4*)&sA[trow * AROW + kc * 32 + koffA];
    a[kc].q[0] = pA[0];   // K offsets +0..+7
    a[kc].q[1] = pA[1];   // K offsets +8..+15
  }

  int tg[8];
#pragma unroll
  for (int r = 0; r < 8; ++r) tg[r] = sTgt[rbase + r];

  float s[8], tl[8];
#pragma unroll
  for (int r = 0; r < 8; ++r) { s[r] = 0.f; tl[r] = -3.4e38f; }

  // Two n-tiles per iteration: nt (tile A) and nt+8 (tile B).
  for (int nt = wid; nt < NN / 16; nt += 16) {
    v8f acc0 = {};
    v8f acc1 = {};
    const unsigned int* base0 = wsB + (((size_t)nt * 8) * 32 + lane) * 8;
    const unsigned int* base1 = wsB + (((size_t)(nt + 8) * 8) * 32 + lane) * 8;
#pragma unroll
    for (int kc = 0; kc < 8; ++kc) {
      Frag16 b0, b1;
      const uint4* pB0 = (const uint4*)(base0 + (size_t)kc * 256);
      const uint4* pB1 = (const uint4*)(base1 + (size_t)kc * 256);
      b0.q[0] = pB0[0]; b0.q[1] = pB0[1];
      b1.q[0] = pB1[0]; b1.q[1] = pB1[1];
      acc0 = __builtin_amdgcn_wmma_f32_16x16x32_bf16(
          false, a[kc].v, false, b0.v, (short)0, acc0, false, false);
      acc1 = __builtin_amdgcn_wmma_f32_16x16x32_bf16(
          false, a[kc].v, false, b1.v, (short)0, acc1, false, false);
    }
    int n0 = nt * 16 + nc;
    int n1 = (nt + 8) * 16 + nc;
#pragma unroll
    for (int r = 0; r < 8; ++r) {
      float lg0 = acc0[r];
      float lg1 = acc1[r];
      s[r] += __expf(lg0) + __expf(lg1);    // logits ~N(0,0.32): no overflow risk
      if (tg[r] == n0) tl[r] = lg0;         // exactly one hit per token in block
      if (tg[r] == n1) tl[r] = lg1;
    }
  }

  // Reduce over the 16 lanes of each half-wave group (same token per r).
#pragma unroll
  for (int off = 1; off <= 8; off <<= 1) {
#pragma unroll
    for (int r = 0; r < 8; ++r) {
      s[r] += __shfl_xor(s[r], off);
      tl[r] = fmaxf(tl[r], __shfl_xor(tl[r], off));
    }
  }
  if (nc == 0) {
#pragma unroll
    for (int r = 0; r < 8; ++r) {
      sS[wid][rbase + r] = s[r];
      sTL[wid][rbase + r] = tl[r];
    }
  }
  __syncthreads();

  if (tid < 32) {
    float contrib = 0.f;
    if (tid < 16) {
      float st = 0.f, tlt = -3.4e38f;
#pragma unroll
      for (int w = 0; w < 8; ++w) {
        st += sS[w][tid];
        tlt = fmaxf(tlt, sTL[w][tid]);
      }
      float ce = logf(st) - tlt;          // = logsumexp - logit[target]
      int gt = t0 + tid;
      if (pad[gt] && mmask[gt]) contrib = ce;
    }
#pragma unroll
    for (int off = 1; off <= 8; off <<= 1) contrib += __shfl_xor(contrib, off);
    if (tid == 0) atomicAdd(ce_sum, contrib);
  }
}

// ---------------------------------------------------------------------------
// Kernel 6: finalize  loss = sum(ce*m) / (sum(m) * C), C = 1
// ---------------------------------------------------------------------------
__global__ void k_final(const unsigned char* __restrict__ pad,
                        const unsigned char* __restrict__ mmask,
                        const float* __restrict__ ce_sum, float* __restrict__ out) {
  __shared__ float red[256];
  float c = 0.f;
  for (int i = threadIdx.x; i < BT; i += 256)
    c += (pad[i] && mmask[i]) ? 1.0f : 0.0f;
  red[threadIdx.x] = c;
  __syncthreads();
  for (int off = 128; off >= 1; off >>= 1) {
    if (threadIdx.x < off) red[threadIdx.x] += red[threadIdx.x + off];
    __syncthreads();
  }
  if (threadIdx.x == 0) {
    float denom = red[0] > 0.f ? red[0] : 1.0f;
    out[0] = ce_sum[0] / denom;
  }
}

// ---------------------------------------------------------------------------
extern "C" void kernel_launch(void* const* d_in, const int* in_sizes, int n_in,
                              void* d_out, int out_size, void* d_ws, size_t ws_size,
                              hipStream_t stream) {
  const float* xs          = (const float*)d_in[0];          // [16,512,256]
  const unsigned char* pad = (const unsigned char*)d_in[1];  // [16,512] bool
  const unsigned char* mm  = (const unsigned char*)d_in[2];  // [16,512] bool
  const float* gam         = (const float*)d_in[3];          // [256]
  const float* bet         = (const float*)d_in[4];          // [256]
  const float* P           = (const float*)d_in[5];          // [256,16]
  const float* emb         = (const float*)d_in[6];          // [1,16,8192]
  const float* W           = (const float*)d_in[7];          // [1,256,8192]
  const float* mask_emb    = (const float*)d_in[8];          // [256]

  char* ws = (char*)d_ws;
  float* ce_sum      = (float*)ws;                                      // 256 B slot
  float* proj        = (float*)(ws + 256);                              // 512 KB
  float* enorm       = (float*)(ws + 256 + 512 * 1024);                 // 32 KB
  int*   tgt         = (int*)(ws + 256 + 512 * 1024 + 32 * 1024);       // 32 KB
  unsigned int* wsB  = (unsigned int*)(ws + 256 + 512 * 1024 + 64 * 1024); // 4 MB

  k_lnproj<<<BT / 8, 256, 0, stream>>>(xs, gam, bet, P, proj, ce_sum);
  k_enorm<<<NN / 256, 256, 0, stream>>>(emb, enorm);
  k_wpack<<<(DD * NN / 2) / 256, 256, 0, stream>>>(W, wsB);
  k_argmin<<<BT / 8, 256, 0, stream>>>(proj, emb, enorm, tgt);
  k_main<<<BT / 16, 256, 0, stream>>>(xs, mm, mask_emb, wsB, tgt, pad, ce_sum);
  k_final<<<1, 256, 0, stream>>>(pad, mm, ce_sum, (float*)d_out);
}
